// GroupQueryAttention_59193239273757
// MI455X (gfx1250) — matrix operand order repaired and checked by reading, not compiled, not run on hardware
//
#include <hip/hip_runtime.h>
#include <hip/hip_bf16.h>

// ---------------------------------------------------------------------------
// GQA attention block for MI455X (gfx1250, wave32, WMMA 16x16x32 f16).
//   out = Attn(RoPE(x@Wq), RoPE(x@Wk), x@Wv) @ Wo     (causal, GQA 32q/8kv)
// GEMMs: 128x128x32 tiles, LDS staged by the Tensor Data Mover (TDM,
// tensor_load_to_lds + s_wait_tensorcnt, double buffered), math on
// v_wmma_f32_16x16x32_f16 with f32 accumulation.
// ---------------------------------------------------------------------------

typedef _Float16 half_t;
typedef _Float16 v16h __attribute__((ext_vector_type(16)));
typedef float    v8f  __attribute__((ext_vector_type(8)));
typedef unsigned int u32x4 __attribute__((ext_vector_type(4)));
typedef int          i32x4 __attribute__((ext_vector_type(4)));
typedef int          i32x8 __attribute__((ext_vector_type(8)));

union FragU { unsigned int u[8]; v16h h; };

#define DIMX  2048
#define HQ    32
#define HKV   8
#define HD    64
#define BB    2
#define TT    2048
#define MTOT  (BB * TT)                 // 4096 tokens
#define NQKV  (HQ*HD + 2*HKV*HD)        // 3072 packed qkv cols
#define GRP   (HQ / HKV)                // 4

static __device__ __forceinline__ v8f zero8() {
  v8f z = {0.f,0.f,0.f,0.f,0.f,0.f,0.f,0.f};
  return z;
}

// ---------------------------------------------------------------------------
// TDM: issue a 2D f16 tile load (tile_d0 x tile_d1) from global into LDS.
// D# per CDNA5 ISA ch.8: group0 {count=1, lds_addr, global_addr, type=2},
// group1 {data_size=2B, pad_enable, pad_interval=16 dwords (64B rows),
// pad_amount=4 dwords (16B)} -> LDS row stride = 80B (40 halves), which kills
// bank conflicts for the fragment gathers.  Tracked by TENSORcnt.
// ---------------------------------------------------------------------------
static __device__ __forceinline__ void tdm_load_tile_f16(
    unsigned lds_byte_off, const void* gptr,
    unsigned tensor_d0, unsigned tensor_d1,
    unsigned tile_d0, unsigned tile_d1, unsigned stride0) {
  unsigned long long ga = (unsigned long long)gptr;
  u32x4 g0 = { 1u,                                   // count=1 (valid user D#)
               lds_byte_off,                         // lds_addr [63:32]
               (unsigned)ga,                         // global_addr[31:0]
               (unsigned)((ga >> 32) & 0x01FFFFFFull) | (2u << 30) }; // type=2
  unsigned dw0 = (1u << 16)    // data_size = 2 bytes
               | (1u << 20)    // pad_enable
               | (3u << 22)    // pad_interval: 16 dwords (=64B tile row)
               | (3u << 25);   // pad_amount: 4 dwords (=16B)
  i32x8 g1 = { (int)dw0,
               (int)((tensor_d0 & 0xFFFFu) << 16),                      // dim0 lo
               (int)((tensor_d0 >> 16) | ((tensor_d1 & 0xFFFFu) << 16)),// dim0 hi|dim1 lo
               (int)((tensor_d1 >> 16) | (tile_d0 << 16)),              // dim1 hi|tile0
               (int)(tile_d1 & 0xFFFFu),                                // tile1 (tile2=0)
               (int)stride0,                                            // dim0 stride lo
               0, 0 };
  i32x4 z4 = {0,0,0,0};
#if defined(__clang_major__) && (__clang_major__ >= 23)
  i32x8 z8 = {0,0,0,0,0,0,0,0};
  __builtin_amdgcn_tensor_load_to_lds(g0, g1, z4, z4, z8, 0);
#else
  __builtin_amdgcn_tensor_load_to_lds(g0, g1, z4, z4, 0);
#endif
}

// ---------------------------------------------------------------------------
// fp32 -> fp16 prep (one-time, bandwidth-trivial). Weights are written
// TRANSPOSED ([N][K]) so GEMM B-tiles are plain row-major 2D tiles for TDM.
// ---------------------------------------------------------------------------
__global__ void cvt_f32_f16(const float* __restrict__ s, half_t* __restrict__ d, int n) {
  int i = blockIdx.x * blockDim.x + threadIdx.x;
  if (i < n) d[i] = (half_t)s[i];
}

__global__ void cvt_f32_f16_T(const float* __restrict__ s, half_t* __restrict__ d,
                              int R, int C) {
  int i = blockIdx.x * blockDim.x + threadIdx.x;
  if (i >= R * C) return;
  int r = i / C, c = i % C;
  d[(size_t)c * R + r] = (half_t)s[i];
}

__global__ void pack_wqkv_T(const float* __restrict__ Wq, const float* __restrict__ Wk,
                            const float* __restrict__ Wv, half_t* __restrict__ out) {
  int i = blockIdx.x * blockDim.x + threadIdx.x;
  if (i >= DIMX * NQKV) return;
  int r = i / NQKV, c = i % NQKV;
  float v;
  if (c < HQ*HD)             v = Wq[(size_t)r * (HQ*HD)  + c];
  else if (c < HQ*HD+HKV*HD) v = Wk[(size_t)r * (HKV*HD) + (c - HQ*HD)];
  else                       v = Wv[(size_t)r * (HKV*HD) + (c - HQ*HD - HKV*HD)];
  out[(size_t)c * DIMX + r] = (half_t)v;                 // transposed [NQKV][DIMX]
}

// ---------------------------------------------------------------------------
// Tiled f16 GEMM: C[M,N] = A[M,K] @ Bt[N,K]^T.  Block tile 128x128, BK=32.
// 256 threads = 8 waves; wave w owns a 32x64 sub-tile (2 A-frags x 4 B-frags,
// 8 WMMA per k-step).  Tiles staged by TDM, double buffered in dynamic LDS:
//   [A0 | B0 | A1 | B1], each 128 rows x 40 halves (padded by TDM).
// ---------------------------------------------------------------------------
template <bool OUT_F16>
__global__ __launch_bounds__(256)
void gemm_f16(const half_t* __restrict__ A, const half_t* __restrict__ Bt,
              void* __restrict__ Cv, int M, int N, int K) {
  extern __shared__ half_t smem[];                 // 4 x 5120 halves = 40 KB
  const int tid  = threadIdx.x;
  const int wave = tid >> 5;
  const int lane = tid & 31;
  const int nloc = lane & 15;
  const int hi   = lane >> 4;
  const int m0   = blockIdx.x * 128;
  const int n0   = blockIdx.y * 128;
  const int rw   = (wave & 3) * 32;                // wave row sub-tile
  const int cw   = (wave >> 2) * 64;               // wave col sub-tile

  v8f acc[2][4];
  #pragma unroll
  for (int a = 0; a < 2; ++a)
    #pragma unroll
    for (int c = 0; c < 4; ++c) acc[a][c] = zero8();

  const int nsteps = K / 32;
  if (wave == 0) {                                 // prologue: fill buffer 0
    tdm_load_tile_f16(0,     A  + (size_t)m0 * K, (unsigned)K, (unsigned)M, 32, 128, (unsigned)K);
    tdm_load_tile_f16(10240, Bt + (size_t)n0 * K, (unsigned)K, (unsigned)N, 32, 128, (unsigned)K);
  }

  for (int s = 0; s < nsteps; ++s) {
    if (wave == 0) {
      if (s + 1 < nsteps) {                        // prefetch next into alt buffer
        const unsigned nb = (unsigned)((s + 1) & 1) * 20480u;
        const int k1 = (s + 1) * 32;
        tdm_load_tile_f16(nb,          A  + (size_t)m0 * K + k1, (unsigned)K, (unsigned)M, 32, 128, (unsigned)K);
        tdm_load_tile_f16(nb + 10240u, Bt + (size_t)n0 * K + k1, (unsigned)K, (unsigned)N, 32, 128, (unsigned)K);
        __builtin_amdgcn_s_wait_tensorcnt(2);      // oldest two (cur buf) done
      } else {
        __builtin_amdgcn_s_wait_tensorcnt(0);
      }
    }
    __syncthreads();                               // cur buffer visible to all

    const half_t* As = smem + (size_t)(s & 1) * 10240;
    const half_t* Bs = As + 5120;

    FragU fa[2], fb[4];
    #pragma unroll
    for (int a = 0; a < 2; ++a) {                  // A layout: hi selects K half
      const int row = rw + a * 16 + nloc;
      #pragma unroll
      for (int j = 0; j < 8; ++j) {
        int kk = hi * 16 + 2 * j;
        fa[a].u[j] = *(const unsigned int*)&As[row * 40 + kk];
      }
    }
    #pragma unroll
    for (int c = 0; c < 4; ++c) {                  // B layout: hi selects K half
      const int row = cw + c * 16 + nloc;
      #pragma unroll
      for (int j = 0; j < 8; ++j)
        fb[c].u[j] = *(const unsigned int*)&Bs[row * 40 + hi * 16 + 2 * j];
    }
    #pragma unroll
    for (int a = 0; a < 2; ++a)
      #pragma unroll
      for (int c = 0; c < 4; ++c)
        acc[a][c] = __builtin_amdgcn_wmma_f32_16x16x32_f16(
            false, fa[a].h, false, fb[c].h, (short)0, acc[a][c], false, false);

    __syncthreads();                               // before cur buffer is reused
  }

  // C layout: VGPR r -> row (r + 8*hi), col nloc
  #pragma unroll
  for (int a = 0; a < 2; ++a)
    #pragma unroll
    for (int c = 0; c < 4; ++c)
      #pragma unroll
      for (int r = 0; r < 8; ++r) {
        int gm = m0 + rw + a * 16 + r + 8 * hi;
        int gn = n0 + cw + c * 16 + nloc;
        float v = acc[a][c][r];
        if (OUT_F16) ((half_t*)Cv)[(size_t)gm * N + gn] = (half_t)v;
        else         ((float*) Cv)[(size_t)gm * N + gn] = v;
      }
}

// ---------------------------------------------------------------------------
// RoPE + bias + head reshape.  qkv is [4096][3072] f16 (packed q|k|v).
//   q -> q_f16 [B][HQ][T][HD]    (rope)
//   k -> k_f16 [B][HKV][T][HD]   (rope)
//   v -> vT    [B][HKV][HD][T]   (transposed so P@V B-frags are contiguous)
// ---------------------------------------------------------------------------
__global__ void rope_reshape(const half_t* __restrict__ qkv,
                             const float* __restrict__ cosb, const float* __restrict__ sinb,
                             const float* __restrict__ bq, const float* __restrict__ bk,
                             const float* __restrict__ bv,
                             half_t* __restrict__ qo, half_t* __restrict__ ko,
                             half_t* __restrict__ vTo) {
  int i = blockIdx.x * blockDim.x + threadIdx.x;
  if (i >= MTOT * NQKV) return;
  int row = i / NQKV;          // b*T + t
  int col = i % NQKV;
  int b = row / TT, t = row % TT;
  int d = col & 63;
  float val = (float)qkv[i];
  float cv = cosb[(size_t)row * HD + d];
  float sv = sinb[(size_t)row * HD + d];

  if (col < HQ * HD) {
    int h = col >> 6;
    val += bq[col];
    float oth = (float)qkv[(d < 32) ? (i + 32) : (i - 32)]
              + bq[(d < 32) ? (col + 32) : (col - 32)];
    float rot = (d < 32) ? -oth : oth;
    qo[(((size_t)b * HQ + h) * TT + t) * HD + d] = (half_t)(val * cv + rot * sv);
  } else if (col < HQ * HD + HKV * HD) {
    int kc = col - HQ * HD;
    int h = kc >> 6;
    val += bk[kc];
    float oth = (float)qkv[(d < 32) ? (i + 32) : (i - 32)]
              + bk[(d < 32) ? (kc + 32) : (kc - 32)];
    float rot = (d < 32) ? -oth : oth;
    ko[(((size_t)b * HKV + h) * TT + t) * HD + d] = (half_t)(val * cv + rot * sv);
  } else {
    int vc = col - HQ * HD - HKV * HD;
    int h = vc >> 6;
    val += bv[vc];
    vTo[(((size_t)b * HKV + h) * HD + d) * TT + t] = (half_t)val;
  }
}

// ---------------------------------------------------------------------------
// Flash attention (causal, GQA).  Grid: (T/64, HQ, B).  128 threads = 4 waves,
// wave w owns 16 q rows.  Per 32-key tile: 4 WMMA (Q.K^T) -> online softmax in
// C-layout registers (shfl_xor row reduce) -> P staged via per-wave LDS ->
// 4 WMMA (P.V).  All waves run identical trip counts (no block barrier).
// K/V stream from L2 (whole working set < 192MB L2).
// ---------------------------------------------------------------------------
__global__ __launch_bounds__(128)
void attn_kernel(const half_t* __restrict__ q, const half_t* __restrict__ k,
                 const half_t* __restrict__ vT, half_t* __restrict__ attn) {
  __shared__ half_t Pl[4][16][32 + 8];

  const int wave = threadIdx.x >> 5;
  const int lane = threadIdx.x & 31;
  const int nloc = lane & 15;
  const int hi   = lane >> 4;
  const int qt   = blockIdx.x;          // 64-row q tile
  const int qh   = blockIdx.y;
  const int b    = blockIdx.z;
  const int kvh  = qh / GRP;
  const int qrow0 = qt * 64 + wave * 16;

  // Q fragments: two K-halves of HD=64
  const half_t* qrow = q + (((size_t)b * HQ + qh) * TT + (qrow0 + nloc)) * HD;
  FragU fq[2];
  #pragma unroll
  for (int f = 0; f < 2; ++f)
    #pragma unroll
    for (int j = 0; j < 8; ++j) {
      int kk = f * 32 + hi * 16 + 2 * j;
      fq[f].u[j] = *(const unsigned int*)&qrow[kk];
    }

  v8f accv[4];
  #pragma unroll
  for (int c = 0; c < 4; ++c) accv[c] = zero8();
  float mrow[8], lrow[8];
  #pragma unroll
  for (int r = 0; r < 8; ++r) { mrow[r] = -1e30f; lrow[r] = 0.f; }

  const half_t* kbase = k  + ((size_t)b * HKV + kvh) * TT * HD;
  const half_t* vbase = vT + ((size_t)b * HKV + kvh) * HD * TT;

  const int nkt = qt * 2 + 2;           // key tiles of 32, up to causal limit
  for (int kt = 0; kt < nkt; ++kt) {
    const int key0 = kt * 32;
    v8f s0 = zero8(), s1 = zero8();
    #pragma unroll
    for (int f = 0; f < 2; ++f) {       // keys [key0, key0+16)
      FragU fk;
      const half_t* kr = kbase + (size_t)(key0 + nloc) * HD + f * 32;
      #pragma unroll
      for (int j = 0; j < 8; ++j) fk.u[j] = *(const unsigned int*)&kr[hi * 16 + 2 * j];
      s0 = __builtin_amdgcn_wmma_f32_16x16x32_f16(false, fq[f].h, false, fk.h,
                                                  (short)0, s0, false, false);
    }
    #pragma unroll
    for (int f = 0; f < 2; ++f) {       // keys [key0+16, key0+32)
      FragU fk;
      const half_t* kr = kbase + (size_t)(key0 + 16 + nloc) * HD + f * 32;
      #pragma unroll
      for (int j = 0; j < 8; ++j) fk.u[j] = *(const unsigned int*)&kr[hi * 16 + 2 * j];
      s1 = __builtin_amdgcn_wmma_f32_16x16x32_f16(false, fq[f].h, false, fk.h,
                                                  (short)0, s1, false, false);
    }

    // online softmax (scale = 1/sqrt(64))
    #pragma unroll
    for (int r = 0; r < 8; ++r) {
      int grow = qrow0 + r + 8 * hi;
      float e0 = s0[r] * 0.125f, e1 = s1[r] * 0.125f;
      if (key0 + nloc > grow)      e0 = -1e30f;
      if (key0 + 16 + nloc > grow) e1 = -1e30f;
      float tmax = fmaxf(e0, e1);
      #pragma unroll
      for (int off = 8; off >= 1; off >>= 1)
        tmax = fmaxf(tmax, __shfl_xor(tmax, off, 16));
      float nm = fmaxf(mrow[r], tmax);
      float p0 = __expf(e0 - nm), p1 = __expf(e1 - nm);
      float ps = p0 + p1;
      #pragma unroll
      for (int off = 8; off >= 1; off >>= 1)
        ps += __shfl_xor(ps, off, 16);
      float alpha = __expf(mrow[r] - nm);
      lrow[r] = lrow[r] * alpha + ps;
      mrow[r] = nm;
      #pragma unroll
      for (int c = 0; c < 4; ++c) accv[c][r] = accv[c][r] * alpha;
      Pl[wave][r + 8 * hi][nloc]      = (half_t)p0;
      Pl[wave][r + 8 * hi][16 + nloc] = (half_t)p1;
    }
    __builtin_amdgcn_wave_barrier();
    asm volatile("s_wait_dscnt 0" ::: "memory");   // wave-local LDS RAW

    // P fragment (A layout, 16x32)
    FragU fp;
    #pragma unroll
    for (int j = 0; j < 8; ++j) {
      int kk = hi * 16 + 2 * j;
      fp.u[j] = *(const unsigned int*)&Pl[wave][nloc][kk];
    }
    // P @ V over the 4 HD column tiles
    #pragma unroll
    for (int c = 0; c < 4; ++c) {
      FragU fv;
      const half_t* vr = vbase + (size_t)(c * 16 + nloc) * TT + key0;
      #pragma unroll
      for (int j = 0; j < 8; ++j) fv.u[j] = *(const unsigned int*)&vr[hi * 16 + 2 * j];
      accv[c] = __builtin_amdgcn_wmma_f32_16x16x32_f16(false, fp.h, false, fv.h,
                                                       (short)0, accv[c], false, false);
    }
    __builtin_amdgcn_wave_barrier();
  }

  // normalize + store to attn_f16 [B*T][HQ*HD]
  #pragma unroll
  for (int c = 0; c < 4; ++c) {
    #pragma unroll
    for (int r = 0; r < 8; ++r) {
      int grow = qrow0 + r + 8 * hi;
      float o = accv[c][r] * (1.0f / lrow[r]);
      attn[((size_t)b * TT + grow) * DIMX + qh * HD + c * 16 + nloc] = (half_t)o;
    }
  }
}

// ---------------------------------------------------------------------------
// Launch.  Inputs: x, cos, sin, Wq, bq, Wk, bk, Wv, bv, Wo.
// ---------------------------------------------------------------------------
extern "C" void kernel_launch(void* const* d_in, const int* in_sizes, int n_in,
                              void* d_out, int out_size, void* d_ws, size_t ws_size,
                              hipStream_t stream) {
  const float* x   = (const float*)d_in[0];
  const float* cs  = (const float*)d_in[1];
  const float* sn  = (const float*)d_in[2];
  const float* Wq  = (const float*)d_in[3];
  const float* bq  = (const float*)d_in[4];
  const float* Wk  = (const float*)d_in[5];
  const float* bk  = (const float*)d_in[6];
  const float* Wv  = (const float*)d_in[7];
  const float* bv  = (const float*)d_in[8];
  const float* Wo  = (const float*)d_in[9];
  float* out = (float*)d_out;

  char* ws = (char*)d_ws;
  half_t* x_f16     = (half_t*)(ws);                       // 16,777,216 B
  half_t* WqkvT_f16 = (half_t*)(ws + 16777216);            // 12,582,912 B  [3072][2048]
  half_t* WoT_f16   = (half_t*)(ws + 29360128);            //  8,388,608 B  [2048][2048]
  half_t* qkv_f16   = (half_t*)(ws + 37748736);            // 25,165,824 B
  half_t* q_f16     = (half_t*)(ws + 62914560);            // 16,777,216 B
  half_t* k_f16     = (half_t*)(ws + 79691776);            //  4,194,304 B
  half_t* vT_f16    = (half_t*)(ws + 83886080);            //  4,194,304 B
  half_t* attn_f16  = (half_t*)(ws + 88080384);            // 16,777,216 B -> 104,857,600

  const unsigned GEMM_LDS = 4u * 128u * 40u * 2u;          // 40,960 B dynamic

  // 1) precision prep (weights transposed for TDM-friendly B tiles)
  cvt_f32_f16<<<(MTOT * DIMX) / 256, 256, 0, stream>>>(x, x_f16, MTOT * DIMX);
  pack_wqkv_T<<<(DIMX * NQKV) / 256, 256, 0, stream>>>(Wq, Wk, Wv, WqkvT_f16);
  cvt_f32_f16_T<<<(DIMX * DIMX) / 256, 256, 0, stream>>>(Wo, WoT_f16, DIMX, DIMX);

  // 2) fused QKV projection: [4096,2048] @ [2048,3072]
  gemm_f16<true><<<dim3(MTOT / 128, NQKV / 128), 256, GEMM_LDS, stream>>>(
      x_f16, WqkvT_f16, (void*)qkv_f16, MTOT, NQKV, DIMX);

  // 3) RoPE + bias + head reshape (+ V transpose)
  rope_reshape<<<(MTOT * NQKV) / 256, 256, 0, stream>>>(
      qkv_f16, cs, sn, bq, bk, bv, q_f16, k_f16, vT_f16);

  // 4) causal GQA flash attention
  attn_kernel<<<dim3(TT / 64, HQ, BB), 128, 0, stream>>>(q_f16, k_f16, vT_f16, attn_f16);

  // 5) output projection: [4096,2048] @ [2048,2048] -> f32 out
  gemm_f16<false><<<dim3(MTOT / 128, DIMX / 128), 256, GEMM_LDS, stream>>>(
      attn_f16, WoT_f16, (void*)out, MTOT, DIMX, DIMX);

  (void)in_sizes; (void)n_in; (void)out_size; (void)ws_size;
}
